// ScannedRNN_72035191488596
// MI455X (gfx1250) — matrix operand order repaired and checked by reading, not compiled, not run on hardware
//
#include <hip/hip_runtime.h>
#include <cstdint>

// ---------------------------------------------------------------------------
// ScannedRNN (GRU with per-step reset) for MI455X / gfx1250, wave32 + WMMA.
//
// T=512, B=128, H=512.  The scan is a true recurrence: 512 dependent steps of
// six [128,512]x[512,512] GEMMs + gates. Latency-bound, not bandwidth-bound
// (x+ys = 268 MB @ 23.3 TB/s ~= 11.5 us; weights = 3 MB bf16, L2-resident).
//
// Strategy:
//  * bf16 WMMA (v_wmma_f32_16x16x32_bf16) for all 6 GEMM streams, f32 acc.
//    Carry h stays f32 for the (1-z)*n + z*h blend (no compounding error).
//  * Weights packed once into exact WMMA B-fragment layout (coalesced b128).
//  * x converted to bf16 once, up front (parallel, off the serial path).
//  * ONE persistent kernel runs all 512 steps: device-side grid barrier via
//    agent-scope atomics removes 512 kernel-launch/drain overheads.
//  * x tile for step t+1 is prefetched into the other LDS buffer with
//    GLOBAL_LOAD_ASYNC_TO_LDS_B128 (ASYNCcnt / s_wait_asynccnt) while step
//    t's WMMAs run (guarded by __has_builtin; falls back to plain loads).
// ---------------------------------------------------------------------------

#define T_STEPS 512
#define BATCH   128
#define HID     512
#define LDSW    520   // padded LDS row stride (bf16): 1040 B -> conflict-free b128
#define NBLK    64    // persistent grid: all co-resident on a many-WGP part

typedef __bf16 v16bf __attribute__((ext_vector_type(16)));
typedef __bf16 v8bf  __attribute__((ext_vector_type(8)));
typedef __bf16 v4bf  __attribute__((ext_vector_type(4)));
typedef float  v8f   __attribute__((ext_vector_type(8)));

struct WeightPtrs { const float* p[6]; };  // Wir, Whr, Wiz, Whz, Win, Whn

static __device__ __forceinline__ v8f wmma_bf16(v16bf a, v16bf b, v8f c) {
  return __builtin_amdgcn_wmma_f32_16x16x32_bf16(false, a, false, b,
                                                 (short)0, c, false, false);
}

// ---- CDNA5 async global->LDS path (guarded; histogram confirms lowering) --
#if __has_builtin(__builtin_amdgcn_global_load_async_to_lds_b128) && \
    __has_builtin(__builtin_amdgcn_s_wait_asynccnt)
#define HAVE_ASYNC 1
typedef int vi4 __attribute__((vector_size(16)));
typedef __attribute__((address_space(1))) vi4 gvi4;   // global (__device__)
typedef __attribute__((address_space(3))) vi4 lvi4;   // LDS (__shared__)
static __device__ __forceinline__ void async_cp16(const void* g, void* l) {
  __builtin_amdgcn_global_load_async_to_lds_b128(
      (gvi4*)(uintptr_t)g,
      (lvi4*)(uint32_t)(uintptr_t)l,   // generic addr low 32b == LDS offset
      0, 0);
}
#else
#define HAVE_ASYNC 0
#endif

// ---------------------------------------------------------------------------
// Pack 6 f32 [512x512] weights into bf16 WMMA B-fragment layout:
//   wpack[((m*32 + nt)*16 + kt)*512 + lane*16 + e] = W_m[k][n]
//   k = kt*32 + (e>=8?16:0) + (lane>=16?8:0) + (e&7) ; n = nt*16 + (lane&15)
// Also zeroes the grid-barrier counter.
// ---------------------------------------------------------------------------
__global__ __launch_bounds__(256)
void pack_weights(__bf16* __restrict__ wpack, WeightPtrs wp,
                  unsigned int* __restrict__ ctr) {
  if (blockIdx.x == 0 && threadIdx.x == 0) *ctr = 0u;
  int flat = blockIdx.x * 256 + threadIdx.x;   // 6*512*512 total
  int e    = flat & 15;
  int lane = (flat >> 4) & 31;
  int kt   = (flat >> 9) & 15;
  int nt   = (flat >> 13) & 31;
  int m    = flat >> 18;
  int k = (kt << 5) + ((e >> 3) << 4) + ((lane >> 4) << 3) + (e & 7);
  int n = (nt << 4) + (lane & 15);
  const float* W = (m == 0) ? wp.p[0] : (m == 1) ? wp.p[1] : (m == 2) ? wp.p[2]
                 : (m == 3) ? wp.p[3] : (m == 4) ? wp.p[4] : wp.p[5];
  wpack[flat] = (__bf16)W[k * HID + n];
}

// Convert all of x to bf16 once (fully parallel; off the serial scan path).
__global__ __launch_bounds__(256)
void pack_x(__bf16* __restrict__ xbf, const float* __restrict__ x) {
  size_t i = (size_t)blockIdx.x * 256 + threadIdx.x;   // float4 index
  float4 v = reinterpret_cast<const float4*>(x)[i];
  v4bf b = { (__bf16)v.x, (__bf16)v.y, (__bf16)v.z, (__bf16)v.w };
  *reinterpret_cast<v4bf*>(xbf + i * 4) = b;
}

// ---------------------------------------------------------------------------
// Persistent scan kernel: NBLK blocks x 128 threads (4 waves) run all steps.
//   block -> row-tile rt = blk&7 (16 batch rows), col-group cg = blk>>3
//   wave  -> col-tile ct = cg*4 + wave (16 hidden cols)
// Step t reads ys[t-1] (h0 at t=0) and writes ys[t]; a device-wide barrier
// (agent-scope atomics on ctr) separates steps.
// ---------------------------------------------------------------------------
__global__ __launch_bounds__(128)
void gru_scan(const __bf16* __restrict__ xbf,          // [T,B,H] bf16
              const unsigned char* __restrict__ rsts,  // [T,B] bool (1B)
              const float* __restrict__ h0,            // [B,H]
              const __bf16* __restrict__ wpack,
              const float* __restrict__ br,
              const float* __restrict__ bz,
              const float* __restrict__ bn,
              float* __restrict__ out,                 // [T,B,H]
              unsigned int* __restrict__ ctr) {
  __shared__ __bf16 xa[2][16][LDSW];   // double-buffered x tile (bf16)
  __shared__ __bf16 ha[16][LDSW];      // reset-masked h tile (bf16)
  __shared__ float  rmask[16];

  const int tid  = threadIdx.x;
  const int rt   = blockIdx.x & 7;
  const int cg   = blockIdx.x >> 3;
  const int b0   = rt << 4;
  const int lane = tid & 31;
  const int wv   = tid >> 5;
  const int ct   = (cg << 2) + wv;
  const int col  = (ct << 4) + (lane & 15);
  const int hi   = lane >> 4;
  const int row  = lane & 15;
  const size_t BH = (size_t)BATCH * HID;

  const float vbr = br[col], vbz = bz[col], vbn = bn[col];

  // Prologue: bring x[0] tile into xa[0] (16 rows x 512 cols bf16 = 16 KB;
  // 1024 chunks of 16 B, 8 per thread).
  {
    const __bf16* xs = xbf;  // t = 0
    #pragma unroll
    for (int j = 0; j < 8; ++j) {
      int i  = tid + (j << 7);
      int r  = i >> 6;
      int c8 = (i & 63) << 3;
#if HAVE_ASYNC
      async_cp16(xs + (b0 + r) * HID + c8, &xa[0][r][c8]);
#else
      *reinterpret_cast<v8bf*>(&xa[0][r][c8]) =
          *reinterpret_cast<const v8bf*>(xs + (b0 + r) * HID + c8);
#endif
    }
  }

  for (int t = 0; t < T_STEPS; ++t) {
    const int buf = t & 1;
    const float* hp = t ? out + (size_t)(t - 1) * BH : h0;
    const unsigned char* rst = rsts + (size_t)t * BATCH;

    // ---- grid barrier: all blocks finished step t-1 (out[t-1] complete) --
    if (t > 0) {
      if (tid == 0) {
        unsigned tgt = (unsigned)(NBLK * t);
        while (__hip_atomic_load(ctr, __ATOMIC_ACQUIRE,
                                 __HIP_MEMORY_SCOPE_AGENT) < tgt)
          __builtin_amdgcn_s_sleep(1);
      }
      __syncthreads();
      __threadfence();   // acquire: invalidate stale lines before reading ys[t-1]
    }

#if HAVE_ASYNC
    __builtin_amdgcn_s_wait_asynccnt(0);   // x[t] tile landed in xa[buf]
#endif

    // ---- stage reset-masked h (f32 -> bf16) and reset mask ----
    if (tid < 16) rmask[tid] = rst[b0 + tid] ? 0.f : 1.f;
    const float4* h4 = reinterpret_cast<const float4*>(hp);
    for (int i = tid; i < 16 * (HID / 4); i += 128) {
      int r  = i >> 7;
      int c4 = (i & 127) << 2;
      int b  = b0 + r;
      float m = rst[b] ? 0.f : 1.f;
      float4 hv = h4[b * (HID / 4) + (c4 >> 2)];
      v4bf hb = { (__bf16)(hv.x * m), (__bf16)(hv.y * m),
                  (__bf16)(hv.z * m), (__bf16)(hv.w * m) };
      *reinterpret_cast<v4bf*>(&ha[r][c4]) = hb;
    }
    __syncthreads();   // xa[buf], ha, rmask visible to all waves

    // ---- prefetch x[t+1] tile into the other buffer (overlaps k-loop) ----
    if (t + 1 < T_STEPS) {
      const __bf16* xs = xbf + (size_t)(t + 1) * BH;
      #pragma unroll
      for (int j = 0; j < 8; ++j) {
        int i  = tid + (j << 7);
        int r  = i >> 6;
        int c8 = (i & 63) << 3;
#if HAVE_ASYNC
        async_cp16(xs + (b0 + r) * HID + c8, &xa[buf ^ 1][r][c8]);
#else
        *reinterpret_cast<v8bf*>(&xa[buf ^ 1][r][c8]) =
            *reinterpret_cast<const v8bf*>(xs + (b0 + r) * HID + c8);
#endif
      }
    }

    // ---- accumulators (C layout: VGPR v -> row M=v+hi*8, col N=lane&15) --
    v8f accr, accz, accxn, acchn;
    for (int i = 0; i < 8; ++i) {
      accr[i] = vbr; accz[i] = vbz; accxn[i] = 0.f; acchn[i] = vbn;
    }

    // ---- K loop: 16 tiles of K=32; 6 WMMAs each ----
    #pragma unroll 4
    for (int kt = 0; kt < 16; ++kt) {
      const __bf16* xb = &xa[buf][row][(kt << 5) + (hi << 3)];
      const __bf16* hb = &ha[row][(kt << 5) + (hi << 3)];
      v8bf xlo = *reinterpret_cast<const v8bf*>(xb);
      v8bf xhi = *reinterpret_cast<const v8bf*>(xb + 16);
      v8bf hlo = *reinterpret_cast<const v8bf*>(hb);
      v8bf hhi = *reinterpret_cast<const v8bf*>(hb + 16);
      v16bf ax = __builtin_shufflevector(xlo, xhi, 0,1,2,3,4,5,6,7,8,9,10,11,12,13,14,15);
      v16bf ah = __builtin_shufflevector(hlo, hhi, 0,1,2,3,4,5,6,7,8,9,10,11,12,13,14,15);

      const int fb = ((ct * 16 + kt) << 9) + (lane << 4);
      const v16bf* w0 = reinterpret_cast<const v16bf*>(wpack + fb);
      const int MS = (32 * 16 * 512) / 16;   // matrix stride in v16bf
      v16bf bir = w0[0];
      v16bf bhr = w0[MS];
      v16bf biz = w0[2 * MS];
      v16bf bhz = w0[3 * MS];
      v16bf bin = w0[4 * MS];
      v16bf bhn = w0[5 * MS];

      if (kt < 15) __builtin_prefetch((const void*)(wpack + fb + 512), 0, 1);

      accr  = wmma_bf16(ax, bir, accr);
      accz  = wmma_bf16(ax, biz, accz);
      accxn = wmma_bf16(ax, bin, accxn);
      acchn = wmma_bf16(ah, bhn, acchn);
      accr  = wmma_bf16(ah, bhr, accr);
      accz  = wmma_bf16(ah, bhz, accz);
    }

    // ---- epilogue: gates + f32 carry blend; write ys[t] ----
    #pragma unroll
    for (int v = 0; v < 8; ++v) {
      int rloc = v + (hi << 3);
      int b    = b0 + rloc;
      float m  = rmask[rloc];
      float hv = hp[b * HID + col] * m;
      float r  = 1.f / (1.f + __expf(-accr[v]));
      float z  = 1.f / (1.f + __expf(-accz[v]));
      float n  = tanhf(accxn[v] + r * acchn[v]);
      out[(size_t)t * BH + b * HID + col] = (1.f - z) * n + z * hv;
    }

    // ---- arrive: make ys[t] agent-visible, then bump the barrier ----
    __threadfence();
    __syncthreads();
    if (tid == 0)
      __hip_atomic_fetch_add(ctr, 1u, __ATOMIC_RELEASE,
                             __HIP_MEMORY_SCOPE_AGENT);
  }
}

// ---------------------------------------------------------------------------
// Fallback (small workspace): one kernel per step, x converted on the fly.
// ---------------------------------------------------------------------------
__global__ __launch_bounds__(128)
void gru_step(const float* __restrict__ xt, const unsigned char* __restrict__ rst,
              const float* __restrict__ hprev, const __bf16* __restrict__ wpack,
              const float* __restrict__ br, const float* __restrict__ bz,
              const float* __restrict__ bn, float* __restrict__ out) {
  __shared__ __bf16 xa[16][LDSW];
  __shared__ __bf16 ha[16][LDSW];
  __shared__ float  rmask[16];
  const int tid = threadIdx.x;
  const int rt = blockIdx.x & 7, cg = blockIdx.x >> 3, b0 = rt << 4;
  if (tid < 16) rmask[tid] = rst[b0 + tid] ? 0.f : 1.f;
  const float4* x4 = reinterpret_cast<const float4*>(xt);
  const float4* h4 = reinterpret_cast<const float4*>(hprev);
  for (int i = tid; i < 16 * (HID / 4); i += 128) {
    int r = i >> 7, c4 = (i & 127) << 2, b = b0 + r;
    float m = rst[b] ? 0.f : 1.f;
    float4 xv = x4[b * (HID / 4) + (c4 >> 2)];
    float4 hv = h4[b * (HID / 4) + (c4 >> 2)];
    v4bf xb = { (__bf16)xv.x, (__bf16)xv.y, (__bf16)xv.z, (__bf16)xv.w };
    v4bf hb = { (__bf16)(hv.x * m), (__bf16)(hv.y * m),
                (__bf16)(hv.z * m), (__bf16)(hv.w * m) };
    *reinterpret_cast<v4bf*>(&xa[r][c4]) = xb;
    *reinterpret_cast<v4bf*>(&ha[r][c4]) = hb;
  }
  __syncthreads();
  const int lane = tid & 31, wv = tid >> 5;
  const int ct = (cg << 2) + wv, col = (ct << 4) + (lane & 15);
  const int hi = lane >> 4, row = lane & 15;
  v8f accr, accz, accxn, acchn;
  {
    float vbr = br[col], vbz = bz[col], vbn = bn[col];
    for (int i = 0; i < 8; ++i) { accr[i]=vbr; accz[i]=vbz; accxn[i]=0.f; acchn[i]=vbn; }
  }
  #pragma unroll 4
  for (int kt = 0; kt < 16; ++kt) {
    const __bf16* xb = &xa[row][(kt << 5) + (hi << 3)];
    const __bf16* hb = &ha[row][(kt << 5) + (hi << 3)];
    v8bf xlo = *reinterpret_cast<const v8bf*>(xb);
    v8bf xhi = *reinterpret_cast<const v8bf*>(xb + 16);
    v8bf hlo = *reinterpret_cast<const v8bf*>(hb);
    v8bf hhi = *reinterpret_cast<const v8bf*>(hb + 16);
    v16bf ax = __builtin_shufflevector(xlo, xhi, 0,1,2,3,4,5,6,7,8,9,10,11,12,13,14,15);
    v16bf ah = __builtin_shufflevector(hlo, hhi, 0,1,2,3,4,5,6,7,8,9,10,11,12,13,14,15);
    const int fb = ((ct * 16 + kt) << 9) + (lane << 4);
    const v16bf* w0 = reinterpret_cast<const v16bf*>(wpack + fb);
    const int MS = (32 * 16 * 512) / 16;
    v16bf bir = w0[0];      v16bf bhr = w0[MS];
    v16bf biz = w0[2 * MS]; v16bf bhz = w0[3 * MS];
    v16bf bin = w0[4 * MS]; v16bf bhn = w0[5 * MS];
    if (kt < 15) __builtin_prefetch((const void*)(wpack + fb + 512), 0, 1);
    accr  = wmma_bf16(ax, bir, accr);
    accz  = wmma_bf16(ax, biz, accz);
    accxn = wmma_bf16(ax, bin, accxn);
    acchn = wmma_bf16(ah, bhn, acchn);
    accr  = wmma_bf16(ah, bhr, accr);
    accz  = wmma_bf16(ah, bhz, accz);
  }
  #pragma unroll
  for (int v = 0; v < 8; ++v) {
    int rloc = v + (hi << 3), b = b0 + rloc;
    float m = rmask[rloc];
    float hv = hprev[b * HID + col] * m;
    float r = 1.f / (1.f + __expf(-accr[v]));
    float z = 1.f / (1.f + __expf(-accz[v]));
    float n = tanhf(accxn[v] + r * acchn[v]);
    out[b * HID + col] = (1.f - z) * n + z * hv;
  }
}

// ---------------------------------------------------------------------------
extern "C" void kernel_launch(void* const* d_in, const int* in_sizes, int n_in,
                              void* d_out, int out_size, void* d_ws, size_t ws_size,
                              hipStream_t stream) {
  (void)in_sizes; (void)n_in; (void)out_size;
  const float* x            = (const float*)d_in[0];
  const unsigned char* rsts = (const unsigned char*)d_in[1];
  const float* h0           = (const float*)d_in[2];
  const float* Wir          = (const float*)d_in[3];
  const float* Wiz          = (const float*)d_in[4];
  const float* Win          = (const float*)d_in[5];
  const float* Whr          = (const float*)d_in[6];
  const float* Whz          = (const float*)d_in[7];
  const float* Whn          = (const float*)d_in[8];
  const float* br           = (const float*)d_in[9];
  const float* bz           = (const float*)d_in[10];
  const float* bn           = (const float*)d_in[11];
  float* out = (float*)d_out;

  const size_t BH     = (size_t)BATCH * HID;
  const size_t WP_SZ  = (size_t)6 * HID * HID * sizeof(__bf16);  // 3 MiB
  const size_t XBF_OF = WP_SZ + 256;                             // ctr at WP_SZ
  const size_t NEED   = XBF_OF + (size_t)T_STEPS * BH * sizeof(__bf16);

  __bf16* wpack     = reinterpret_cast<__bf16*>(d_ws);
  unsigned int* ctr = reinterpret_cast<unsigned int*>((char*)d_ws + WP_SZ);
  __bf16* xbf       = reinterpret_cast<__bf16*>((char*)d_ws + XBF_OF);

  WeightPtrs wp;
  wp.p[0] = Wir; wp.p[1] = Whr; wp.p[2] = Wiz;
  wp.p[3] = Whz; wp.p[4] = Win; wp.p[5] = Whn;
  pack_weights<<<6 * HID * HID / 256, 256, 0, stream>>>(wpack, wp, ctr);

  if (ws_size >= NEED) {
    // Persistent path: pre-convert x, then one resident kernel scans all T.
    pack_x<<<(int)((size_t)T_STEPS * BH / 4 / 256), 256, 0, stream>>>(xbf, x);
    gru_scan<<<NBLK, 128, 0, stream>>>(xbf, rsts, h0, wpack, br, bz, bn,
                                       out, ctr);
  } else {
    // Small-workspace fallback: one kernel per step (stream-ordered scan).
    for (int t = 0; t < T_STEPS; ++t) {
      const float* hp = (t == 0) ? h0 : out + (size_t)(t - 1) * BH;
      gru_step<<<64, 128, 0, stream>>>(x + (size_t)t * BH,
                                       rsts + (size_t)t * BATCH,
                                       hp, wpack, br, bz, bn,
                                       out + (size_t)t * BH);
    }
  }
}
